// DPCNN_13159779795241
// MI455X (gfx1250) — matrix operand order repaired and checked by reading, not compiled, not run on hardware
//
#include <hip/hip_runtime.h>
#include <hip/hip_bf16.h>
#include <math.h>

// ---------------------------------------------------------------------------
// DPCNN forward on MI455X (gfx1250, wave32, WMMA + TDM).
// Convs = implicit GEMM, bf16 inputs, fp32 accumulation via
// v_wmma_f32_16x16x32_bf16. Activations kept position-major (B, rows, C) in
// bf16 so each conv's im2col K-run (3*Cin) is contiguous memory.
// Weight panels are DMA'd global->LDS with tensor_load_to_lds (double
// buffered, s_wait_tensorcnt), using TDM LDS padding to avoid bank conflicts.
// Each 256-thread block: 8 waves x (64ch x 16pos) tiles sharing one panel.
// ---------------------------------------------------------------------------

typedef __attribute__((ext_vector_type(16))) __bf16 v16bf;
typedef __attribute__((ext_vector_type(8)))  __bf16 v8bf;
typedef __attribute__((ext_vector_type(8)))  float  v8f;
typedef __attribute__((ext_vector_type(4)))  unsigned int u32x4;
typedef __attribute__((ext_vector_type(8)))  unsigned int u32x8;

#define BATCH   64
#define SEQ     512
#define EDIM    768
#define CDIM    256
#define PADTOK  1
#define NCLS    10
#define XB_ROWS 272    // bf16 block-phase buffer row allocation (max Lp+2=257)
#define PX_ROWS 256    // fp32 pooled buffer row allocation (max Lp = 255)

#define KCHUNK      128                  // K elems per TDM weight chunk
#define LDS_ROW_B   (KCHUNK * 2 + 16)    // 272 B: TDM pad -> bank-conflict free
#define LDS_BUF_B   (64 * LDS_ROW_B)     // one 64-row panel chunk (17408 B)

// ---------------- embedding gather + pad mask -> bf16 (B, 512, 768) --------
__global__ __launch_bounds__(256) void embed_kernel(
    const int* __restrict__ ids, const float* __restrict__ table,
    __bf16* __restrict__ out)
{
    int bl = blockIdx.x;                       // 0 .. B*SEQ-1
    int id = ids[bl];
    float m = (id != PADTOK) ? 1.0f : 0.0f;
    const float* row = table + (size_t)id * EDIM;
    __bf16* o = out + (size_t)bl * EDIM;
    for (int e = threadIdx.x; e < EDIM; e += 256)
        o[e] = (__bf16)(row[e] * m);
}

// ---------------- weight repacks: [c][e][k] -> bf16 [c][k*E + e] ------------
__global__ __launch_bounds__(256) void repack_region(
    const float* __restrict__ w, __bf16* __restrict__ wr)
{
    int c = blockIdx.x;
    for (int i = threadIdx.x; i < 3 * EDIM; i += 256) {
        int k = i / EDIM, e = i % EDIM;
        wr[(size_t)c * (3 * EDIM) + i] = (__bf16)w[((size_t)c * EDIM + e) * 3 + k];
    }
}
__global__ __launch_bounds__(256) void repack_conv(
    const float* __restrict__ w, __bf16* __restrict__ wc)
{
    int c = blockIdx.x;
    for (int i = threadIdx.x; i < 3 * CDIM; i += 256) {
        int k = i / CDIM, ci = i % CDIM;
        wc[(size_t)c * (3 * CDIM) + i] = (__bf16)w[((size_t)c * CDIM + ci) * 3 + k];
    }
}

// ---------------- TDM: 2D weight chunk (KCHUNK x 64 rows) global -> LDS -----
// D# per CDNA5 ISA 8.3/8.4. data_size=2B, tile = 128 elems x 64 rows,
// tensor_dim0_stride = KK, LDS pad: every 64 DWORDs (one 256B row) add
// 4 DWORDs (16B) -> LDS row stride 272B (bank-conflict-free fragment reads).
__device__ __forceinline__ void tdm_load_weights(
    const __bf16* gsrc, unsigned lds_off, int KK)
{
    unsigned long long ga = (unsigned long long)(size_t)gsrc;
    u32x4 g0;
    g0[0] = 1u;                                            // count=1 (valid)
    g0[1] = lds_off;                                       // lds_addr
    g0[2] = (unsigned)ga;                                  // global_addr lo
    g0[3] = (unsigned)((ga >> 32) & 0x01FFFFFFu)           // global_addr hi
          | 0x80000000u;                                   // type=2 (image)
    u32x8 g1;
    g1[0] = (1u << 16)        // data_size = 2 bytes
          | (1u << 20)        // pad_enable
          | (5u << 22)        // pad_interval: 64 DWORDs
          | (3u << 25);       // pad_amount:   4 DWORDs
    g1[1] = ((unsigned)KK & 0xffffu) << 16;                // tensor_dim0 lo16
    g1[2] = ((unsigned)KK >> 16) | (64u << 16);            // dim0 hi | dim1 lo
    g1[3] = ((unsigned)KCHUNK << 16);                      // tile_dim0
    g1[4] = 64u;                                           // tile_dim1
    g1[5] = (unsigned)KK;                                  // dim0_stride lo32
    g1[6] = 0u;
    g1[7] = 0u;
    asm volatile("tensor_load_to_lds %0, %1" :: "s"(g0), "s"(g1) : "memory");
}

// ---------------- WMMA implicit-GEMM conv (k=3 folded into K dim) -----------
// Block = 8 waves. Wave w -> 64ch x 16pos tile at positions
// (blockIdx.x*8 + w)*16. Weight panel (64 x KK) streamed through LDS in
// double-buffered KCHUNK slices by wave 0 via TDM.
// Fragment layouts (wave32, contiguous k order, same for A and B):
//   A lane(lr,half): elems i -> K=k0+half*16+i (two 16B loads)
//   B lane(lr,half): col=lr, elems i -> K=k0+half*16+i (one 32B load)
//   D elem v of acc t: channel = cg*64 + t*16 + half*8 + v, pos = tile*16 + lr
__global__ __launch_bounds__(256) void conv_wmma_kernel(
    const __bf16* __restrict__ X, int rowC, int xrows,     // (B, xrows, rowC)
    const __bf16* __restrict__ W, int KK,                  // (256, KK)
    const float*  __restrict__ bias,
    int Lout,
    float*  __restrict__ outF, int orow_off, int orows,    // optional fp32 out
    __bf16* __restrict__ outB, int brow_off, int brows,    // optional bf16 relu
    const float* __restrict__ resid, int rrows)            // optional residual
{
    __shared__ __align__(16) char smem[2 * LDS_BUF_B];

    const int tid   = threadIdx.x;
    const int wv    = tid >> 5;
    const int lane  = tid & 31;
    const int lr    = lane & 15;
    const int half  = lane >> 4;
    const int b     = blockIdx.z;
    const int cg    = blockIdx.y;               // 0..3 : 64-channel group
    const int lt    = blockIdx.x * 8 + wv;      // this wave's position tile
    const int pos   = lt * 16 + lr;
    const int posc  = (pos < Lout) ? pos : (Lout - 1);

    const __bf16* xb    = X + ((size_t)b * xrows + posc) * rowC + half * 16;
    const __bf16* wbase = W + (size_t)(cg * 64) * KK;
    const int nchunks   = KK / KCHUNK;

    v8f acc[4];
    #pragma unroll
    for (int t = 0; t < 4; ++t) acc[t] = (v8f){};

    if (tid < 32)   // wave 0 drives the TDM pipeline
        tdm_load_weights(wbase, (unsigned)(size_t)&smem[0], KK);

    for (int i = 0; i < nchunks; ++i) {
        if (tid < 32) {
            if (i + 1 < nchunks) {
                tdm_load_weights(wbase + (i + 1) * KCHUNK,
                                 (unsigned)(size_t)&smem[((i + 1) & 1) * LDS_BUF_B],
                                 KK);
                __builtin_amdgcn_s_wait_tensorcnt(1);  // chunk i landed
            } else {
                __builtin_amdgcn_s_wait_tensorcnt(0);
            }
        }
        __syncthreads();

        const char* buf = &smem[(i & 1) * LDS_BUF_B];
        const int   kg  = i * KCHUNK;
        #pragma unroll
        for (int k0c = 0; k0c < KCHUNK; k0c += 32) {
            __builtin_prefetch(xb + kg + k0c + 512, 0, 0);   // global_prefetch_b8
            v16bf bf = *(const v16bf*)(xb + kg + k0c);
            #pragma unroll
            for (int t = 0; t < 4; ++t) {
                const char* ap = buf + (t * 16 + lr) * LDS_ROW_B
                                     + (k0c + half * 16) * 2;
                v8bf lo = *(const v8bf*)(ap);
                v8bf hi = *(const v8bf*)(ap + 16);   // +8 elems
                v16bf a = __builtin_shufflevector(lo, hi,
                        0, 1, 2, 3, 4, 5, 6, 7, 8, 9, 10, 11, 12, 13, 14, 15);
                acc[t] = __builtin_amdgcn_wmma_f32_16x16x32_bf16(
                             false, a, false, bf, (short)0, acc[t], false, false);
            }
        }
        __syncthreads();    // all waves done with buf before TDM overwrites it
    }

    if (pos < Lout) {
        #pragma unroll
        for (int t = 0; t < 4; ++t) {
            #pragma unroll
            for (int v = 0; v < 8; ++v) {
                int ch = cg * 64 + t * 16 + half * 8 + v;
                float val = acc[t][v] + bias[ch];
                if (resid) val += resid[((size_t)b * rrows + pos) * CDIM + ch];
                if (outF)  outF[((size_t)b * orows + orow_off + pos) * CDIM + ch] = val;
                if (outB)  outB[((size_t)b * brows + brow_off + pos) * CDIM + ch] =
                               (__bf16)fmaxf(val, 0.0f);
            }
        }
    }
}

// ---------------- zero the two pad rows of an fp32 (B, rows, 256) buffer ----
__global__ __launch_bounds__(256) void zero_pad_rows(
    float* __restrict__ x, int rows, int r0, int r1)
{
    int b = blockIdx.x, c = threadIdx.x;
    x[((size_t)b * rows + r0) * CDIM + c] = 0.0f;
    x[((size_t)b * rows + r1) * CDIM + c] = 0.0f;
}

// ---------------- BN (train-mode, stats include pad rows) -------------------
__global__ void init_sums(float* __restrict__ s) { s[threadIdx.x] = 0.0f; }

__global__ __launch_bounds__(256) void bn_stats(
    const float* __restrict__ x, float* __restrict__ sums)
{
    int c = threadIdx.x;
    size_t r0 = (size_t)blockIdx.x * 256;        // 128 blocks x 256 rows
    float s = 0.0f, q = 0.0f;
    for (int r = 0; r < 256; ++r) {
        float v = x[(r0 + r) * CDIM + c];
        s += v; q += v * v;
    }
    atomicAdd(&sums[c], s);
    atomicAdd(&sums[CDIM + c], q);
}

__global__ __launch_bounds__(256) void bn_apply_relu(
    const float* __restrict__ x, const float* __restrict__ sums,
    const float* __restrict__ g, const float* __restrict__ be,
    __bf16* __restrict__ out)
{
    int c = threadIdx.x;
    size_t row = blockIdx.x;                     // 0 .. B*SEQ-1
    const float invN = 1.0f / (float)(BATCH * SEQ);
    float mean = sums[c] * invN;
    float var  = sums[CDIM + c] * invN - mean * mean;
    float rs   = rsqrtf(var + 1e-5f);
    float v = (x[row * CDIM + c] - mean) * rs * g[c] + be[c];
    out[row * CDIM + c] = (__bf16)fmaxf(v, 0.0f);
}

// ---------------- block prologue: maxpool3 s2 (+right zero pad) -------------
__global__ __launch_bounds__(256) void pool_kernel(
    const float* __restrict__ x, int xrows, int L, int Lp,
    float* __restrict__ px, int prows,
    __bf16* __restrict__ pxb, __bf16* __restrict__ y1b, int brows)
{
    int c = threadIdx.x;
    int p = blockIdx.x, b = blockIdx.y;
    const float* xb = x + (size_t)b * xrows * CDIM;
    float v0 = xb[(size_t)(2 * p)     * CDIM + c];
    float v1 = xb[(size_t)(2 * p + 1) * CDIM + c];
    float v2 = (2 * p + 2 < L) ? xb[(size_t)(2 * p + 2) * CDIM + c] : 0.0f;
    float mx = fmaxf(v0, fmaxf(v1, v2));
    px[((size_t)b * prows + p) * CDIM + c] = mx;
    pxb[((size_t)b * brows + 1 + p) * CDIM + c] = (__bf16)fmaxf(mx, 0.0f);
    if (p == 0) {
        pxb[((size_t)b * brows) * CDIM + c] = (__bf16)0.0f;
        y1b[((size_t)b * brows) * CDIM + c] = (__bf16)0.0f;
    }
    if (p == Lp - 1) {
        pxb[((size_t)b * brows + Lp + 1) * CDIM + c] = (__bf16)0.0f;
        y1b[((size_t)b * brows + Lp + 1) * CDIM + c] = (__bf16)0.0f;
    }
}

// ---------------- head: select head, GEMV, log-softmax, NLL mean ------------
__global__ __launch_bounds__(64) void head_kernel(
    const float* __restrict__ xfin, int xrows,
    const float* __restrict__ groups, const int* __restrict__ labels,
    const float* __restrict__ w_heads, const float* __restrict__ b_heads,
    float* __restrict__ out)
{
    __shared__ float acc[BATCH];
    int b = threadIdx.x;
    const float* gp = groups + b * 6;
    int idx = 0; float best = gp[0];
    for (int j = 1; j < 6; ++j) if (gp[j] > best) { best = gp[j]; idx = j; }
    if (idx > 4) idx = 4;
    const float* feat = xfin + (size_t)b * xrows * CDIM;   // position 0
    float logits[NCLS];
    #pragma unroll
    for (int j = 0; j < NCLS; ++j) logits[j] = b_heads[idx * NCLS + j];
    for (int c = 0; c < CDIM; ++c) {
        float f = feat[c];
        const float* wp = w_heads + ((size_t)idx * CDIM + c) * NCLS;
        #pragma unroll
        for (int j = 0; j < NCLS; ++j) logits[j] += f * wp[j];
    }
    float mx = logits[0];
    #pragma unroll
    for (int j = 1; j < NCLS; ++j) mx = fmaxf(mx, logits[j]);
    float s = 0.0f;
    #pragma unroll
    for (int j = 0; j < NCLS; ++j) s += expf(logits[j] - mx);
    float lse = mx + logf(s);
    acc[b] = -(logits[labels[b]] - lse);
    __syncthreads();
    if (b == 0) {
        float t = 0.0f;
        for (int i = 0; i < BATCH; ++i) t += acc[i];
        out[0] = t / (float)BATCH;
    }
}

// ---------------------------------------------------------------------------
extern "C" void kernel_launch(void* const* d_in, const int* in_sizes, int n_in,
                              void* d_out, int out_size, void* d_ws, size_t ws_size,
                              hipStream_t stream)
{
    (void)in_sizes; (void)n_in; (void)out_size; (void)ws_size;
    const int*   input_ids = (const int*)  d_in[0];
    const float* groups    = (const float*)d_in[1];
    const int*   labels    = (const int*)  d_in[2];
    const float* emb_table = (const float*)d_in[3];
    const float* w_region  = (const float*)d_in[4];
    const float* b_region  = (const float*)d_in[5];
    const float* w_conv    = (const float*)d_in[6];
    const float* b_conv    = (const float*)d_in[7];
    const float* g1  = (const float*)d_in[8];
    const float* be1 = (const float*)d_in[9];
    const float* g2  = (const float*)d_in[10];
    const float* be2 = (const float*)d_in[11];
    const float* w_heads = (const float*)d_in[12];
    const float* b_heads = (const float*)d_in[13];
    float* out = (float*)d_out;

    // ---- workspace carve-up (all 256B aligned) ----
    char* ws = (char*)d_ws;
    size_t off = 0;
    auto take = [&](size_t bytes) -> char* {
        char* p = ws + off;
        off = (off + bytes + 255) & ~(size_t)255;
        return p;
    };
    __bf16* WR   = (__bf16*)take((size_t)CDIM * 3 * EDIM * 2);          // 1.13 MB
    __bf16* WC   = (__bf16*)take((size_t)CDIM * 3 * CDIM * 2);          // 384 KB
    float*  SUMS = (float*) take(2 * CDIM * sizeof(float));
    char*   XEMB = take((size_t)BATCH * SEQ * EDIM * 2);                // 48 MB
    float*  ABUF = (float*) take((size_t)BATCH * SEQ * CDIM * 4);       // 32 MB
    float*  BBUF = (float*) take((size_t)BATCH * SEQ * CDIM * 4);       // 32 MB
    __bf16* ACT  = (__bf16*)take((size_t)BATCH * SEQ * CDIM * 2);       // 16 MB
    // block-phase buffers overlay XEMB (free after the region conv)
    float*  PX  = (float*)XEMB;                                         // 16 MB
    __bf16* PXB = (__bf16*)(XEMB + (size_t)BATCH * PX_ROWS * CDIM * 4); // 8.9 MB
    __bf16* Y1B = PXB + (size_t)BATCH * XB_ROWS * CDIM;                 // 8.9 MB

    // ---- weight repacks + embedding ----
    repack_region<<<CDIM, 256, 0, stream>>>(w_region, WR);
    repack_conv  <<<CDIM, 256, 0, stream>>>(w_conv,   WC);
    embed_kernel <<<BATCH * SEQ, 256, 0, stream>>>(input_ids, emb_table,
                                                   (__bf16*)XEMB);

    // ---- region conv (K=2304) -> ABUF rows 1..510, pads zero ----
    zero_pad_rows<<<BATCH, 256, 0, stream>>>(ABUF, SEQ, 0, SEQ - 1);
    conv_wmma_kernel<<<dim3((510 + 127) / 128, 4, BATCH), 256, 0, stream>>>(
        (__bf16*)XEMB, EDIM, SEQ, WR, 3 * EDIM, b_region, 510,
        ABUF, 1, SEQ, nullptr, 0, 0, nullptr, 0);

    // ---- BN1 + relu -> ACT ----
    init_sums<<<1, 2 * CDIM, 0, stream>>>(SUMS);
    bn_stats<<<(BATCH * SEQ) / 256, 256, 0, stream>>>(ABUF, SUMS);
    bn_apply_relu<<<BATCH * SEQ, 256, 0, stream>>>(ABUF, SUMS, g1, be1, ACT);

    // ---- conv (K=768) -> BBUF rows 1..510 ----
    zero_pad_rows<<<BATCH, 256, 0, stream>>>(BBUF, SEQ, 0, SEQ - 1);
    conv_wmma_kernel<<<dim3((510 + 127) / 128, 4, BATCH), 256, 0, stream>>>(
        ACT, CDIM, SEQ, WC, 3 * CDIM, b_conv, 510,
        BBUF, 1, SEQ, nullptr, 0, 0, nullptr, 0);

    // ---- BN2 + relu -> ACT ----
    init_sums<<<1, 2 * CDIM, 0, stream>>>(SUMS);
    bn_stats<<<(BATCH * SEQ) / 256, 256, 0, stream>>>(BBUF, SUMS);
    bn_apply_relu<<<BATCH * SEQ, 256, 0, stream>>>(BBUF, SUMS, g2, be2, ACT);

    // ---- conv (K=768) -> ABUF rows 0..509 : x entering pyramid, L=510 ----
    conv_wmma_kernel<<<dim3((510 + 127) / 128, 4, BATCH), 256, 0, stream>>>(
        ACT, CDIM, SEQ, WC, 3 * CDIM, b_conv, 510,
        ABUF, 0, SEQ, nullptr, 0, 0, nullptr, 0);

    // ---- pyramid blocks: 510->255->127->63->31->15->7->3->1 ----
    float* xcur = ABUF;
    float* xnext = BBUF;
    int L = 510;
    while (L > 2) {
        int Lp = (L - 2) / 2 + 1;
        pool_kernel<<<dim3(Lp, BATCH), 256, 0, stream>>>(
            xcur, SEQ, L, Lp, PX, PX_ROWS, PXB, Y1B, XB_ROWS);
        // conv1: relu(pad(px)) * W -> relu'd bf16 into Y1B rows 1..Lp
        conv_wmma_kernel<<<dim3((Lp + 127) / 128, 4, BATCH), 256, 0, stream>>>(
            PXB, CDIM, XB_ROWS, WC, 3 * CDIM, b_conv, Lp,
            nullptr, 0, 0, Y1B, 1, XB_ROWS, nullptr, 0);
        // conv2: + px residual -> fp32 xnext rows 0..Lp-1
        conv_wmma_kernel<<<dim3((Lp + 127) / 128, 4, BATCH), 256, 0, stream>>>(
            Y1B, CDIM, XB_ROWS, WC, 3 * CDIM, b_conv, Lp,
            xnext, 0, SEQ, nullptr, 0, 0, PX, PX_ROWS);
        float* t = xcur; xcur = xnext; xnext = t;
        L = Lp;
    }

    // ---- head: select, GEMV, log-softmax, NLL mean -> out[0] ----
    head_kernel<<<1, BATCH, 0, stream>>>(xcur, SEQ, groups, labels,
                                         w_heads, b_heads, out);
}
